// MultiHeadAttention_83614423319314
// MI455X (gfx1250) — matrix operand order repaired and checked by reading, not compiled, not run on hardware
//
#include <hip/hip_runtime.h>
#include <hip/hip_bf16.h>

// ---------------------------------------------------------------------------
// MI455X (gfx1250) multi-head attention forward.
// Matmuls on v_wmma_f32_16x16x32_bf16; attention K-tiles staged by the
// Tensor Data Mover (tensor_load_to_lds + s_wait_tensorcnt).
// ---------------------------------------------------------------------------

typedef __attribute__((ext_vector_type(16))) __bf16 v16bf;
typedef __attribute__((ext_vector_type(8)))  float  v8f;
typedef __attribute__((ext_vector_type(4)))  unsigned int u32x4;
typedef __attribute__((ext_vector_type(8)))  unsigned int u32x8;

union BFrag { v16bf v; unsigned short u[16]; uint4 q[2]; };

__device__ __forceinline__ unsigned short f2bf(float f) {
  union { float f; unsigned u; } a; a.f = f;
  unsigned x = a.u;
  return (unsigned short)((x + 0x7FFFu + ((x >> 16) & 1u)) >> 16);  // RNE
}
__device__ __forceinline__ float bf2f(unsigned short h) {
  union { unsigned u; float f; } a; a.u = ((unsigned)h) << 16; return a.f;
}

// A fragment (16x32 bf16): lane holds row = lane%16, K = (lane/16)*8 + {0..7,16..23}
__device__ __forceinline__ v16bf frag_a(const unsigned short* lds, int row, int stride, int lane) {
  int kb = (lane >> 4) << 3;
  BFrag f;
  f.q[0] = *(const uint4*)(lds + row * stride + kb);
  f.q[1] = *(const uint4*)(lds + row * stride + kb + 16);
  return f.v;
}
// B fragment (32x16 bf16): lane holds col = lane%16, K = kofs + (lane/16)*8 + {0..7,16..23}
__device__ __forceinline__ v16bf frag_b(const unsigned short* lds, int col, int stride, int kofs, int lane) {
  int kb = kofs + ((lane >> 4) << 3);
  BFrag f;
  f.q[0] = *(const uint4*)(lds + col * stride + kb);
  f.q[1] = *(const uint4*)(lds + col * stride + kb + 16);
  return f.v;
}

#define WMMA_BF16(a, b, c) \
  __builtin_amdgcn_wmma_f32_16x16x32_bf16(false, (a), false, (b), (short)0, (c), false, false)

#define ZERO8 {0.f,0.f,0.f,0.f,0.f,0.f,0.f,0.f}

// ---------------------------------------------------------------------------
// Kernel 1: qkv = x @ Wqkv + bqkv ; scatter bf16 into Q/K/V [B,H,T,D]
// M=8192, N=2304, K=768. Macro-tile 128x128, K-step 32, 8 waves (4x2).
// ---------------------------------------------------------------------------
__global__ __launch_bounds__(256) void qkv_gemm_kernel(
    const float* __restrict__ x, const float* __restrict__ W,
    const float* __restrict__ bias,
    unsigned short* __restrict__ qbuf, unsigned short* __restrict__ kbuf,
    unsigned short* __restrict__ vbuf) {
  const int K = 768, N = 2304;
  __shared__ unsigned short As[128 * 40];
  __shared__ unsigned short Bs[128 * 40];
  int tid = threadIdx.x, lane = tid & 31, wave = tid >> 5;
  int wm = wave & 3, wn = wave >> 2;          // wave tile: 32 rows x 64 cols
  int m0 = blockIdx.y * 128, n0 = blockIdx.x * 128;

  v8f acc[2][4];
#pragma unroll
  for (int i = 0; i < 2; ++i)
#pragma unroll
    for (int j = 0; j < 4; ++j) acc[i][j] = (v8f)ZERO8;

  for (int k0 = 0; k0 < K; k0 += 32) {
    // A tile: 128x32 f32 -> bf16, row-major stride 40
#pragma unroll
    for (int i = 0; i < 4; ++i) {
      int row = (tid >> 3) + i * 32;
      int c4 = (tid & 7) * 4;
      const float4 f = *(const float4*)(x + (size_t)(m0 + row) * K + k0 + c4);
      unsigned p0 = (unsigned)f2bf(f.x) | ((unsigned)f2bf(f.y) << 16);
      unsigned p1 = (unsigned)f2bf(f.z) | ((unsigned)f2bf(f.w) << 16);
      *(unsigned*)(As + row * 40 + c4)     = p0;
      *(unsigned*)(As + row * 40 + c4 + 2) = p1;
    }
    // B tile: 32x128 f32 -> bf16 transposed [n][k], stride 40
#pragma unroll
    for (int i = 0; i < 4; ++i) {
      int kr = (tid >> 5) + i * 8;
      int n4 = (tid & 31) * 4;
      const float4 f = *(const float4*)(W + (size_t)(k0 + kr) * N + n0 + n4);
      Bs[(n4 + 0) * 40 + kr] = f2bf(f.x);
      Bs[(n4 + 1) * 40 + kr] = f2bf(f.y);
      Bs[(n4 + 2) * 40 + kr] = f2bf(f.z);
      Bs[(n4 + 3) * 40 + kr] = f2bf(f.w);
    }
    // prefetch next K-slice toward the WGP while WMMAs run
    if (k0 + 32 < K) {
      __builtin_prefetch(x + (size_t)(m0 + (tid >> 3)) * K + k0 + 32 + (tid & 7) * 4, 0, 1);
      __builtin_prefetch(W + (size_t)(k0 + 32 + (tid >> 5)) * N + n0 + (tid & 31) * 4, 0, 1);
    }
    __syncthreads();
    v16bf af[2], bf[4];
#pragma unroll
    for (int mt = 0; mt < 2; ++mt)
      af[mt] = frag_a(As, wm * 32 + mt * 16 + (lane & 15), 40, lane);
#pragma unroll
    for (int nt = 0; nt < 4; ++nt)
      bf[nt] = frag_b(Bs, wn * 64 + nt * 16 + (lane & 15), 40, 0, lane);
#pragma unroll
    for (int mt = 0; mt < 2; ++mt)
#pragma unroll
      for (int nt = 0; nt < 4; ++nt)
        acc[mt][nt] = WMMA_BF16(af[mt], bf[nt], acc[mt][nt]);
    __syncthreads();
  }

  // Epilogue: +bias, scatter bf16 to q/k/v in [B,H,T,D]
#pragma unroll
  for (int mt = 0; mt < 2; ++mt) {
#pragma unroll
    for (int nt = 0; nt < 4; ++nt) {
      int j = n0 + wn * 64 + nt * 16 + (lane & 15);
      float bj = bias[j];
      int mat = j / 768, rem = j - mat * 768;
      int h = rem >> 6, d = rem & 63;
      unsigned short* dst = (mat == 0) ? qbuf : ((mat == 1) ? kbuf : vbuf);
#pragma unroll
      for (int r = 0; r < 8; ++r) {
        int mrow = m0 + wm * 32 + mt * 16 + r + ((lane >> 4) << 3);
        int b = mrow >> 12, t = mrow & 4095;
        dst[(size_t)(((b * 12 + h) * 4096) + t) * 64 + d] = f2bf(acc[mt][nt][r] + bj);
      }
    }
  }
}

// ---------------------------------------------------------------------------
// Kernel 2: flash attention. Block = (b,h) x 128 query rows, 8 waves x 16 rows.
// K tiles via Tensor Data Mover (async DMA into padded LDS); V staged
// transposed by VALU; online softmax; WMMA for S=QK^T and O+=P@V.
// ---------------------------------------------------------------------------
__global__ __launch_bounds__(256) void attn_kernel(
    const unsigned short* __restrict__ Q, const unsigned short* __restrict__ Kx,
    const unsigned short* __restrict__ V, unsigned short* __restrict__ Aout) {
  const int T = 4096, D = 64, H = 12;
  __shared__ unsigned short Ks[32 * 72];       // [key][d], stride 72 (144B rows via TDM pad)
  __shared__ unsigned short Vs[64 * 40];       // [d][key], stride 40
  __shared__ unsigned short Ps[8][16 * 40];    // per-wave P tile [row][key], stride 40
  int tid = threadIdx.x, lane = tid & 31, wave = tid >> 5;
  int bh = blockIdx.x;
  int b = bh / H, h = bh - b * H;
  const unsigned short* Qb = Q + (size_t)bh * T * D;
  const unsigned short* Kb = Kx + (size_t)bh * T * D;
  const unsigned short* Vb = V + (size_t)bh * T * D;
  int qrow0 = blockIdx.y * 128 + wave * 16;
  int half = (lane >> 4) << 3;

  // Q fragments (16x64), pre-scaled by 1/sqrt(D)=0.125
  v16bf qf[2];
  {
    int row = lane & 15, kb0 = (lane >> 4) << 3;
#pragma unroll
    for (int s = 0; s < 2; ++s) {
      const unsigned short* src = Qb + (size_t)(qrow0 + row) * D + s * 32 + kb0;
      BFrag raw, f;
      raw.q[0] = *(const uint4*)(src);
      raw.q[1] = *(const uint4*)(src + 16);
#pragma unroll
      for (int e = 0; e < 16; ++e) f.u[e] = f2bf(bf2f(raw.u[e]) * 0.125f);
      qf[s] = f.v;
    }
  }

  // TDM descriptor group1 (loop-invariant): 2D tile 64(d) x 32(key), 2B elems,
  // LDS pad 4 DWORDs (16B) after every 32 DWORDs (128B row) -> 144B row stride.
  u32x8 g1;
  g1[0] = (1u << 16) | (1u << 20) | (4u << 22) | (3u << 25);
  g1[1] = 64u << 16;            // tensor_dim0[15:0] = 64
  g1[2] = 4096u << 16;          // tensor_dim0 hi = 0, tensor_dim1[15:0] = 4096
  g1[3] = 64u << 16;            // tensor_dim1 hi = 0, tile_dim0 = 64
  g1[4] = 32u;                  // tile_dim1 = 32, tile_dim2 = 0
  g1[5] = 64u;                  // tensor_dim0_stride = 64
  g1[6] = 0u;
  g1[7] = 0u;
  unsigned ldsK = (unsigned)(size_t)(&Ks[0]);   // low 32 bits of flat LDS addr = LDS offset

  float mrow[8], lrow[8];
  v8f o[4];
#pragma unroll
  for (int r = 0; r < 8; ++r) { mrow[r] = -1e30f; lrow[r] = 0.f; }
#pragma unroll
  for (int nd = 0; nd < 4; ++nd) o[nd] = (v8f)ZERO8;

  int nKB = blockIdx.y * 4 + 4;  // causal: keys up to qbase+128
  for (int kb = 0; kb < nKB; ++kb) {
    __syncthreads();
    // Issue K-tile DMA from one wave (TDM is a per-wave op, EXEC-ignored)
    if (wave == 0) {
      unsigned long long ga = (unsigned long long)(size_t)(Kb + (size_t)kb * 32 * D);
      u32x4 g0;
      g0[0] = 1u;                                   // count = 1 valid descriptor
      g0[1] = ldsK;                                 // lds_addr
      g0[2] = (unsigned)ga;                         // global_addr[31:0]
      g0[3] = (unsigned)(ga >> 32) | (2u << 30);    // global_addr[56:32] | type=2
      asm volatile("tensor_load_to_lds %0, %1" :: "s"(g0), "s"(g1) : "memory");
    }
    // V tile staged transposed ([d][key]) by all waves, overlapping the DMA
    {
      int key = tid >> 3, ch = (tid & 7) * 8;
      BFrag vv;
      vv.q[0] = *(const uint4*)(Vb + (size_t)(kb * 32 + key) * D + ch);
#pragma unroll
      for (int e = 0; e < 8; ++e) Vs[(ch + e) * 40 + key] = vv.u[e];
    }
    if (wave == 0) __builtin_amdgcn_s_wait_tensorcnt(0);
    __syncthreads();

    // S = Q @ K^T  (16 rows x 32 keys, two 16-wide subtiles)
    v8f s[2];
#pragma unroll
    for (int nt = 0; nt < 2; ++nt) {
      v8f a = (v8f)ZERO8;
      v16bf b0 = frag_b(Ks, nt * 16 + (lane & 15), 72, 0, lane);
      a = WMMA_BF16(qf[0], b0, a);
      v16bf b1 = frag_b(Ks, nt * 16 + (lane & 15), 72, 32, lane);
      a = WMMA_BF16(qf[1], b1, a);
      s[nt] = a;
    }

    // causal mask
#pragma unroll
    for (int nt = 0; nt < 2; ++nt) {
      int kc = kb * 32 + nt * 16 + (lane & 15);
#pragma unroll
      for (int r = 0; r < 8; ++r)
        if (kc > qrow0 + r + half) s[nt][r] = -1e30f;
    }

    // online softmax (row = r + half; reduce over 16-lane half-wave)
#pragma unroll
    for (int r = 0; r < 8; ++r) {
      float mb = fmaxf(s[0][r], s[1][r]);
#pragma unroll
      for (int off = 8; off >= 1; off >>= 1) mb = fmaxf(mb, __shfl_xor(mb, off, 32));
      float mn = fmaxf(mrow[r], mb);
      float sc = __expf(mrow[r] - mn);
      float p0 = __expf(s[0][r] - mn);
      float p1 = __expf(s[1][r] - mn);
      s[0][r] = p0; s[1][r] = p1;
      float rs = p0 + p1;
#pragma unroll
      for (int off = 8; off >= 1; off >>= 1) rs += __shfl_xor(rs, off, 32);
      lrow[r] = lrow[r] * sc + rs;
      mrow[r] = mn;
#pragma unroll
      for (int nd = 0; nd < 4; ++nd) o[nd][r] *= sc;
    }

    // P (C layout) -> per-wave LDS -> A fragment layout
    unsigned short* pw = &Ps[wave][0];
#pragma unroll
    for (int nt = 0; nt < 2; ++nt)
#pragma unroll
      for (int r = 0; r < 8; ++r)
        pw[(r + half) * 40 + nt * 16 + (lane & 15)] = f2bf(s[nt][r]);
    __asm__ volatile("s_wait_dscnt 0" ::: "memory");
    v16bf pf = frag_a(pw, lane & 15, 40, lane);

    // O += P @ V
#pragma unroll
    for (int nd = 0; nd < 4; ++nd) {
      v16bf vf = frag_b(Vs, nd * 16 + (lane & 15), 40, 0, lane);
      o[nd] = WMMA_BF16(pf, vf, o[nd]);
    }
  }

  // normalize and store bf16 to [B,T,C] for the out projection
#pragma unroll
  for (int nd = 0; nd < 4; ++nd) {
    int d = nd * 16 + (lane & 15);
#pragma unroll
    for (int r = 0; r < 8; ++r) {
      int qr = qrow0 + r + half;
      float val = o[nd][r] / lrow[r];
      Aout[(size_t)(b * 4096 + qr) * 768 + h * 64 + d] = f2bf(val);
    }
  }
}

// ---------------------------------------------------------------------------
// Kernel 3: out = attn(bf16) @ Wout + bout.  M=8192, N=768, K=768.
// ---------------------------------------------------------------------------
__global__ __launch_bounds__(256) void proj_gemm_kernel(
    const unsigned short* __restrict__ A, const float* __restrict__ W,
    const float* __restrict__ bias, float* __restrict__ out) {
  const int K = 768, N = 768;
  __shared__ unsigned short As[128 * 40];
  __shared__ unsigned short Bs[128 * 40];
  int tid = threadIdx.x, lane = tid & 31, wave = tid >> 5;
  int wm = wave & 3, wn = wave >> 2;
  int m0 = blockIdx.y * 128, n0 = blockIdx.x * 128;

  v8f acc[2][4];
#pragma unroll
  for (int i = 0; i < 2; ++i)
#pragma unroll
    for (int j = 0; j < 4; ++j) acc[i][j] = (v8f)ZERO8;

  for (int k0 = 0; k0 < K; k0 += 32) {
    // A tile (already bf16): 128x32, 512 uint4 chunks
#pragma unroll
    for (int i = 0; i < 2; ++i) {
      int c = tid + i * 256;
      int row = c >> 2, ch = (c & 3) * 8;
      *(uint4*)(As + row * 40 + ch) =
          *(const uint4*)(A + (size_t)(m0 + row) * K + k0 + ch);
    }
    // B tile: f32 -> bf16 transposed
#pragma unroll
    for (int i = 0; i < 4; ++i) {
      int kr = (tid >> 5) + i * 8;
      int n4 = (tid & 31) * 4;
      const float4 f = *(const float4*)(W + (size_t)(k0 + kr) * N + n0 + n4);
      Bs[(n4 + 0) * 40 + kr] = f2bf(f.x);
      Bs[(n4 + 1) * 40 + kr] = f2bf(f.y);
      Bs[(n4 + 2) * 40 + kr] = f2bf(f.z);
      Bs[(n4 + 3) * 40 + kr] = f2bf(f.w);
    }
    if (k0 + 32 < K) {
      __builtin_prefetch(A + (size_t)(m0 + (tid >> 2)) * K + k0 + 32 + (tid & 3) * 8, 0, 1);
      __builtin_prefetch(W + (size_t)(k0 + 32 + (tid >> 5)) * N + n0 + (tid & 31) * 4, 0, 1);
    }
    __syncthreads();
    v16bf af[2], bf[4];
#pragma unroll
    for (int mt = 0; mt < 2; ++mt)
      af[mt] = frag_a(As, wm * 32 + mt * 16 + (lane & 15), 40, lane);
#pragma unroll
    for (int nt = 0; nt < 4; ++nt)
      bf[nt] = frag_b(Bs, wn * 64 + nt * 16 + (lane & 15), 40, 0, lane);
#pragma unroll
    for (int mt = 0; mt < 2; ++mt)
#pragma unroll
      for (int nt = 0; nt < 4; ++nt)
        acc[mt][nt] = WMMA_BF16(af[mt], bf[nt], acc[mt][nt]);
    __syncthreads();
  }

#pragma unroll
  for (int mt = 0; mt < 2; ++mt) {
#pragma unroll
    for (int nt = 0; nt < 4; ++nt) {
      int j = n0 + wn * 64 + nt * 16 + (lane & 15);
      float bj = bias[j];
#pragma unroll
      for (int r = 0; r < 8; ++r) {
        int mrow = m0 + wm * 32 + mt * 16 + r + ((lane >> 4) << 3);
        out[(size_t)mrow * N + j] = acc[mt][nt][r] + bj;
      }
    }
  }
}

// ---------------------------------------------------------------------------
extern "C" void kernel_launch(void* const* d_in, const int* in_sizes, int n_in,
                              void* d_out, int out_size, void* d_ws, size_t ws_size,
                              hipStream_t stream) {
  const float* x    = (const float*)d_in[0];
  const float* Wqkv = (const float*)d_in[1];
  const float* bqkv = (const float*)d_in[2];
  const float* Wout = (const float*)d_in[3];
  const float* bout = (const float*)d_in[4];
  float* out = (float*)d_out;

  // Workspace: bf16 Q/K/V in [B,H,T,D] + bf16 attention output in [B,T,C]
  unsigned short* ws = (unsigned short*)d_ws;
  const size_t perBuf = (size_t)2 * 12 * 4096 * 64;  // 6,291,456 elems each
  unsigned short* qb = ws;
  unsigned short* kb = ws + perBuf;
  unsigned short* vb = ws + 2 * perBuf;
  unsigned short* ab = ws + 3 * perBuf;

  qkv_gemm_kernel<<<dim3(18, 64), 256, 0, stream>>>(x, Wqkv, bqkv, qb, kb, vb);
  attn_kernel<<<dim3(24, 32), 256, 0, stream>>>(qb, kb, vb, ab);
  proj_gemm_kernel<<<dim3(6, 64), 256, 0, stream>>>(ab, Wout, bout, out);
}
